// BasicMGU_53386443489965
// MI455X (gfx1250) — matrix operand order repaired and checked by reading, not compiled, not run on hardware
//
#include <hip/hip_runtime.h>
#include <math.h>

// ---- problem dims ----
#define BB 64
#define TT 1024
#define DD 512
#define UU 512
#define PITCH 520   // LDS row pitch in halves: 1040B = 4 banks apart per row -> conflict-free

typedef __attribute__((ext_vector_type(16))) _Float16 v16h;
typedef __attribute__((ext_vector_type(8)))  _Float16 v8h;
typedef __attribute__((ext_vector_type(8)))  float    v8f;
typedef __attribute__((ext_vector_type(4)))  _Float16 h4;
typedef __attribute__((ext_vector_type(4)))  float    f4;

__device__ __forceinline__ v8f zero8() {
  v8f z = {0.f,0.f,0.f,0.f,0.f,0.f,0.f,0.f};
  return z;
}

// A-matrix 16x32 f16 tile (ISA 7.12.2): lane&15 = row; lanes 0-15 hold K {0..7,16..23},
// lanes 16-31 hold K {8..15,24..31}. Two 16B loads per lane.
__device__ __forceinline__ v16h load_tileA(const _Float16* base, int ld, int lane) {
  const int r   = lane & 15;
  const int klo = (lane >> 4) << 3;            // 0 or 8
  const _Float16* p = base + r * ld + klo;
  v8h lo = *(const v8h*)(p);                   // K = klo .. klo+7
  v8h hi = *(const v8h*)(p + 16);              // K = klo+16 .. klo+23
  return __builtin_shufflevector(lo, hi, 0,1,2,3,4,5,6,7,8,9,10,11,12,13,14,15);
}

// B-matrix 32x16 f16 tile from W^T stored row-major [N x K]: lane&15 = column n;
// lanes 0-15 hold K {0..7,16..23}, lanes 16-31 hold K {8..15,24..31}.
__device__ __forceinline__ v16h load_tileB(const _Float16* base, int ld, int lane) {
  const int n   = lane & 15;
  const int klo = (lane >> 4) << 3;            // 0 or 8
  const _Float16* p = base + n * ld + klo;
  v8h lo = *(const v8h*)(p);
  v8h hi = *(const v8h*)(p + 16);
  return __builtin_shufflevector(lo, hi, 0,1,2,3,4,5,6,7,8,9,10,11,12,13,14,15);
}

__device__ __forceinline__ v8f wmma16(v16h a, v16h b, v8f c) {
  // (neg_a, A, neg_b, B, c_mod, C, reuse_a, reuse_b)
  return __builtin_amdgcn_wmma_f32_16x16x32_f16(false, a, false, b, (short)0, c, false, false);
}

__device__ __forceinline__ float fast_sigmoid(float x) {
  return 1.0f / (1.0f + __expf(-x));
}
__device__ __forceinline__ float fast_tanh(float x) {
  float a = __expf(-2.0f * fabsf(x));
  float t = (1.0f - a) / (1.0f + a);
  return copysignf(t, x);
}

// Monotonic grid barrier over `nwg` workgroups (agent scope),
// plus a cluster barrier (NOP when not dispatched as a cluster, ISA 3.2/6.6).
__device__ __forceinline__ void grid_barrier(unsigned* bar, unsigned nwg, unsigned* target) {
  __syncthreads();
  __builtin_amdgcn_s_cluster_barrier();
  if (threadIdx.x == 0) {
    *target += nwg;
    __hip_atomic_fetch_add(bar, 1u, __ATOMIC_RELEASE, __HIP_MEMORY_SCOPE_AGENT);
    while (__hip_atomic_load(bar, __ATOMIC_ACQUIRE, __HIP_MEMORY_SCOPE_AGENT) < *target) {
      __builtin_amdgcn_s_sleep(2);
    }
  }
  __syncthreads();
}

// ---- init: zero h (f32, lives in d_out), h16, hf16, barrier ----
__global__ void mgu_init(float* __restrict__ h, _Float16* __restrict__ h16,
                         _Float16* __restrict__ hf16, unsigned* __restrict__ bar) {
  int i = blockIdx.x * blockDim.x + threadIdx.x;
  if (i < BB * UU) {
    h[i]    = 0.0f;
    h16[i]  = (_Float16)0.0f;
    hf16[i] = (_Float16)0.0f;
  }
  if (i == 0) *bar = 0u;
}

// ---- convert x: f32 -> f16 (vectorized 4-wide) ----
__global__ void mgu_cvt_x(const float* __restrict__ x, _Float16* __restrict__ x16, long n4) {
  long i = (long)blockIdx.x * blockDim.x + threadIdx.x;
  long stride = (long)gridDim.x * blockDim.x;
  for (; i < n4; i += stride) {
    f4 v = ((const f4*)x)[i];
    h4 o;
    o[0] = (_Float16)v[0]; o[1] = (_Float16)v[1];
    o[2] = (_Float16)v[2]; o[3] = (_Float16)v[3];
    ((h4*)x16)[i] = o;
  }
}

// ---- transpose+convert a 512x512 f32 weight into f16 W^T [N x K] ----
__global__ void mgu_tw(const float* __restrict__ w, _Float16* __restrict__ wt) {
  int i = blockIdx.x * blockDim.x + threadIdx.x;   // i = n*512 + k, writes coalesced
  if (i >= 512 * 512) return;
  int k = i & 511;
  int n = i >> 9;
  wt[i] = (_Float16)w[k * 512 + n];
}

// ---- phase 1: xz = x @ W_k, xh = x @ W_u  (M=65536, K=512, N=512) ----
// WG = 8 waves, all share one 64-row slice of WkT/WuT staged in LDS (130KB).
// A tile is register double-buffered so WMMAs only wait on next-iteration loads.
// Outputs are written TIME-MAJOR [T,B,U] as f16 for the scan's streaming reads.
__global__ void __launch_bounds__(256) mgu_gemm_in(
    const _Float16* __restrict__ x16,
    const _Float16* __restrict__ WkT, const _Float16* __restrict__ WuT,
    _Float16* __restrict__ xz, _Float16* __restrict__ xh) {
  extern __shared__ char smem[];
  _Float16* ldsk = (_Float16*)smem;            // 64 rows x PITCH
  _Float16* ldsu = ldsk + 64 * PITCH;

  const int lane = threadIdx.x & 31;
  const int wave = threadIdx.x >> 5;
  const int mblk = blockIdx.x >> 3;            // 0..511
  const int ng   = blockIdx.x & 7;             // 0..7 (group of 4 n-tiles)
  const long m0  = ((long)mblk * 8 + wave) * 16;
  const int  nb  = ng * 64;                    // base row of W^T slice

  // cooperative LDS fill of both weight slices (64 rows x 512 halves each)
  for (int c = threadIdx.x; c < 64 * 64; c += 256) {
    int row = c >> 6;
    int off = (c & 63) * 8;
    *(v8h*)(ldsk + row * PITCH + off) = *(const v8h*)(WkT + (long)(nb + row) * DD + off);
    *(v8h*)(ldsu + row * PITCH + off) = *(const v8h*)(WuT + (long)(nb + row) * DD + off);
  }
  __syncthreads();

  v8f accz[4], acch[4];
#pragma unroll
  for (int j = 0; j < 4; ++j) { accz[j] = zero8(); acch[j] = zero8(); }

  const _Float16* abase = x16 + m0 * DD;
  v16h a_c = load_tileA(abase, DD, lane);
#pragma unroll
  for (int kk = 0; kk < 16; ++kk) {
    v16h a_n = a_c;
    if (kk < 15) a_n = load_tileA(abase + (kk + 1) * 32, DD, lane);  // prefetch next A
#pragma unroll
    for (int j = 0; j < 4; ++j) {
      v16h bk = load_tileB(ldsk + (j * 16) * PITCH + kk * 32, PITCH, lane);
      v16h bu = load_tileB(ldsu + (j * 16) * PITCH + kk * 32, PITCH, lane);
      accz[j] = wmma16(a_c, bk, accz[j]);
      acch[j] = wmma16(a_c, bu, acch[j]);
    }
    a_c = a_n;
  }

  const int dn = lane & 15;
  const int mb = (lane >> 4) << 3;
#pragma unroll
  for (int j = 0; j < 4; ++j) {
    const long n0 = (long)(nb + j * 16);
#pragma unroll
    for (int r = 0; r < 8; ++r) {
      long m  = m0 + mb + r;        // global row = b*T + t
      long b  = m >> 10;            // /TT
      long tt = m & (TT - 1);
      long idx = (tt * BB + b) * UU + n0 + dn;   // time-major [T,B,U]
      xz[idx] = (_Float16)accz[j][r];
      xh[idx] = (_Float16)acch[j][r];
    }
  }
}

// ---- phase 2: persistent scan. 16 WGs x 8 waves = 128 waves = 4x32 tiles of [64,512].
// LDS per WG: both weight slices for its 8 waves (260KB) + one shared A-stage (16.6KB).
// h_old lives in registers for the whole scan; only h16/hf16 round-trip through L2.
// xz/xh are time-major f16: each step reads/prefetches one contiguous 64KB block per array.
__global__ void __launch_bounds__(256) mgu_scan(
    const _Float16* __restrict__ xz, const _Float16* __restrict__ xhg,
    const _Float16* __restrict__ WrT, const _Float16* __restrict__ WurT,
    const float* __restrict__ b_r, const float* __restrict__ b_ur,
    _Float16* __restrict__ h16, _Float16* __restrict__ hf16,
    float* __restrict__ hout, unsigned* __restrict__ bar) {
  extern __shared__ char smem[];
  _Float16* lds_wr  = (_Float16*)smem;                 // 8 waves * 16 rows * PITCH
  _Float16* lds_wur = lds_wr  + 8 * 16 * PITCH;
  _Float16* lds_a   = lds_wur + 8 * 16 * PITCH;        // 16 rows * PITCH (shared stage)

  const int lane = threadIdx.x & 31;
  const int wave = threadIdx.x >> 5;
  const int w    = (blockIdx.x << 3) + wave;           // 0..127
  const int m0   = (w >> 5) * 16;                      // batch-row block (same for all waves in WG)
  const int n0   = (w & 31) * 16;
  const int dn   = n0 + (lane & 15);
  const int mb   = m0 + ((lane >> 4) << 3);
  const float brv  = b_r[dn];
  const float burv = b_ur[dn];

  // one-time: each wave stages its own 16-row weight slices into LDS
  {
    const int row = lane & 15;
    const int seg = (lane >> 4) * 256;                 // halves
    _Float16* dr = lds_wr  + wave * 16 * PITCH + row * PITCH + seg;
    _Float16* du = lds_wur + wave * 16 * PITCH + row * PITCH + seg;
    const _Float16* sr = WrT  + (long)(n0 + row) * UU + seg;
    const _Float16* su = WurT + (long)(n0 + row) * UU + seg;
#pragma unroll 8
    for (int i = 0; i < 32; ++i) {
      *(v8h*)(dr + i * 8) = *(const v8h*)(sr + i * 8);
      *(v8h*)(du + i * 8) = *(const v8h*)(su + i * 8);
    }
  }
  const _Float16* bs1 = lds_wr  + wave * 16 * PITCH;
  const _Float16* bs2 = lds_wur + wave * 16 * PITCH;

  float ho[8];
#pragma unroll
  for (int r = 0; r < 8; ++r) ho[r] = 0.0f;            // h0 = 0 (matches zeroed h16)

  unsigned target = 0;
  for (int t = 0; t < TT; ++t) {
    // prefetch next timestep's contiguous [B,U] blocks into GL2
    if (t + 1 < TT) {
      __builtin_prefetch(&xz [((long)(t + 1) * BB + mb) * UU + dn], 0, 1);
      __builtin_prefetch(&xhg[((long)(t + 1) * BB + mb) * UU + dn], 0, 1);
    }

    // stage h16 rows m0..m0+15 into LDS (cooperative, 16KB)
    for (int c = (int)threadIdx.x; c < 1024; c += 256) {
      int row = c >> 6;
      int off = (c & 63) * 8;
      *(v8h*)(lds_a + row * PITCH + off) = *(const v8h*)(h16 + (m0 + row) * UU + off);
    }
    __syncthreads();

    // f_pre = h @ W_r   (pure LDS + WMMA)
    v8f acc = zero8();
#pragma unroll
    for (int kk = 0; kk < 16; ++kk)
      acc = wmma16(load_tileA(lds_a + kk * 32, PITCH, lane),
                   load_tileB(bs1   + kk * 32, PITCH, lane), acc);

    float fr[8];
#pragma unroll
    for (int r = 0; r < 8; ++r) {
      const int m = mb + r;
      float z = acc[r] + (float)xz[((long)t * BB + m) * UU + dn] + brv;
      fr[r] = fast_sigmoid(z);
      hf16[m * UU + dn] = (_Float16)(ho[r] * fr[r]);   // h*f for GEMM2
    }
    grid_barrier(bar, gridDim.x, &target);

    // stage hf16 rows into the same LDS buffer
    for (int c = (int)threadIdx.x; c < 1024; c += 256) {
      int row = c >> 6;
      int off = (c & 63) * 8;
      *(v8h*)(lds_a + row * PITCH + off) = *(const v8h*)(hf16 + (m0 + row) * UU + off);
    }
    __syncthreads();

    // cand_pre = (h*f) @ W_ur
    v8f acc2 = zero8();
#pragma unroll
    for (int kk = 0; kk < 16; ++kk)
      acc2 = wmma16(load_tileA(lds_a + kk * 32, PITCH, lane),
                    load_tileB(bs2   + kk * 32, PITCH, lane), acc2);

#pragma unroll
    for (int r = 0; r < 8; ++r) {
      const int m = mb + r;
      float g  = acc2[r] + (float)xhg[((long)t * BB + m) * UU + dn] + burv;
      float hc = fast_tanh(g);
      ho[r] = ho[r] + fr[r] * (hc - ho[r]);            // (1-f)h + f*hc
      h16[m * UU + dn] = (_Float16)ho[r];
    }
    grid_barrier(bar, gridDim.x, &target);
  }

  // write final hidden state once
#pragma unroll
  for (int r = 0; r < 8; ++r)
    hout[(mb + r) * UU + dn] = ho[r];
}

extern "C" void kernel_launch(void* const* d_in, const int* in_sizes, int n_in,
                              void* d_out, int out_size, void* d_ws, size_t ws_size,
                              hipStream_t stream) {
  const float* x    = (const float*)d_in[0];  // [B,T,D]
  const float* W_k  = (const float*)d_in[1];  // [D,U]
  const float* W_r  = (const float*)d_in[2];  // [U,U]
  const float* b_r  = (const float*)d_in[3];  // [U]
  const float* W_u  = (const float*)d_in[4];  // [D,U]
  const float* W_ur = (const float*)d_in[5];  // [U,U]
  const float* b_ur = (const float*)d_in[6];  // [U]
  float* hout = (float*)d_out;                // [B,U]

  constexpr size_t SZ_X16 = (size_t)BB * TT * DD * sizeof(_Float16);
  constexpr size_t SZ_XZ  = (size_t)BB * TT * UU * sizeof(_Float16);  // f16, time-major
  constexpr size_t SZ_W   = (size_t)512 * 512 * sizeof(_Float16);
  constexpr size_t SZ_H16 = (size_t)BB * UU * sizeof(_Float16);

  char* ws = (char*)d_ws;
  size_t off = 0;
  _Float16* x16  = (_Float16*)(ws + off); off += SZ_X16;
  _Float16* xz   = (_Float16*)(ws + off); off += SZ_XZ;
  _Float16* xhg  = (_Float16*)(ws + off); off += SZ_XZ;
  _Float16* WkT  = (_Float16*)(ws + off); off += SZ_W;
  _Float16* WuT  = (_Float16*)(ws + off); off += SZ_W;
  _Float16* WrT  = (_Float16*)(ws + off); off += SZ_W;
  _Float16* WurT = (_Float16*)(ws + off); off += SZ_W;
  _Float16* h16  = (_Float16*)(ws + off); off += SZ_H16;
  _Float16* hf16 = (_Float16*)(ws + off); off += SZ_H16;
  unsigned* bar  = (unsigned*)(ws + off); off += 128;

  const size_t smem_gemm = (size_t)2 * 64 * PITCH * sizeof(_Float16);            // 133,120 B
  const size_t smem_scan = (size_t)(2 * 8 * 16 + 16) * PITCH * sizeof(_Float16); // 282,880 B

  mgu_init<<<(BB * UU + 255) / 256, 256, 0, stream>>>(hout, h16, hf16, bar);
  mgu_cvt_x<<<8192, 256, 0, stream>>>(x, x16, (long)BB * TT * DD / 4);
  mgu_tw<<<1024, 256, 0, stream>>>(W_k,  WkT);
  mgu_tw<<<1024, 256, 0, stream>>>(W_u,  WuT);
  mgu_tw<<<1024, 256, 0, stream>>>(W_r,  WrT);
  mgu_tw<<<1024, 256, 0, stream>>>(W_ur, WurT);
  mgu_gemm_in<<<4096, 256, smem_gemm, stream>>>(x16, WkT, WuT, xz, xhg);
  mgu_scan<<<16, 256, smem_scan, stream>>>(xz, xhg, WrT, WurT, b_r, b_ur,
                                           h16, hf16, hout, bar);
}
